// RNNBlock_8959301780017
// MI455X (gfx1250) — compile-verified, matrix operand order checked by reading, not run on hardware
//
#include <hip/hip_runtime.h>

// ---------------------------------------------------------------------------
// Problem constants (from reference): SEQ=784, BS=256, D_H=512, 3 layers.
// M = SEQ*BS = 200704 rows, K = N = 512.
// ---------------------------------------------------------------------------
#define SEQ_LEN   784
#define BATCH     256
#define DH        512
#define MROWS     (SEQ_LEN * BATCH)          // 200704
#define LAYERS    3

typedef __bf16 v16bf __attribute__((ext_vector_type(16)));
typedef float  v8f   __attribute__((ext_vector_type(8)));

struct Pair { uint4 lo, hi; };

__device__ __forceinline__ v16bf make_frag(uint4 lo, uint4 hi) {
    Pair p{lo, hi};
    return __builtin_bit_cast(v16bf, p);
}

__device__ __forceinline__ unsigned short f32_to_bf16(float f) {
    unsigned int u = __float_as_uint(f);
    unsigned int r = u + 0x7FFFu + ((u >> 16) & 1u);   // round-to-nearest-even
    return (unsigned short)(r >> 16);
}

__device__ __forceinline__ float bf16_to_f32(unsigned short h) {
    return __uint_as_float(((unsigned int)h) << 16);
}

// Generic pointer to __shared__ -> LDS byte offset (flat LDS addr low 32 bits)
__device__ __forceinline__ unsigned lds_off(const void* p) {
    return (unsigned)(unsigned long long)p;
}

// ---------------------------------------------------------------------------
// Kernel 1: transpose + convert recurrence weights B,C (3 each, 512x512 f32,
// layout [k][n]) into bf16 WT[mat][n*512 + k], mat = 2*l + (is_C).
// ---------------------------------------------------------------------------
__launch_bounds__(256)
__global__ void wprep_kernel(const float* __restrict__ B,
                             const float* __restrict__ C,
                             unsigned short* __restrict__ wt) {
    int gid = blockIdx.x * 256 + threadIdx.x;       // 6 * 512 * 512 total
    int mat = gid >> 18;                            // 0..5
    int rem = gid & 262143;
    int n = rem >> 9;
    int k = rem & 511;
    int l = mat >> 1;
    const float* src = (mat & 1) ? C : B;
    wt[gid] = f32_to_bf16(src[l * DH * DH + k * DH + n]);
}

// ---------------------------------------------------------------------------
// Kernel 2: h0 = relu(x * W0 + b0)  (outer product, D_IN = 1) -> bf16 act
// ---------------------------------------------------------------------------
__launch_bounds__(256)
__global__ void inproj_kernel(const float* __restrict__ x,
                              const float* __restrict__ W0,
                              const float* __restrict__ b0,
                              unsigned short* __restrict__ act) {
    int gid = blockIdx.x * 256 + threadIdx.x;       // MROWS * DH
    int m = gid >> 9;
    int h = gid & 511;
    float v = x[m] * W0[h] + b0[h];
    act[gid] = f32_to_bf16(v > 0.f ? v : 0.f);
}

// ---------------------------------------------------------------------------
// Kernel 3: bf16 WMMA GEMM  Out[M,512] = A[M,512] @ W[512,512]
//   W given pre-transposed as WT[n][k] bf16.
//   Block tile 128x128, 256 threads = 8 waves (4 m-waves x 2 n-waves),
//   per wave 2x4 tiles of v_wmma_f32_16x16x32_bf16, K-loop of 16 steps.
//   Global->LDS staging uses CDNA5 async-to-LDS loads (ASYNCcnt) with
//   double-buffered LDS so step i+1 loads overlap step i WMMAs.
//   do_relu: clamp at zero before bf16 store (fused activation).
// ---------------------------------------------------------------------------
__launch_bounds__(256)
__global__ void gemm_bf16_kernel(const unsigned short* __restrict__ A,
                                 const unsigned short* __restrict__ WT,
                                 unsigned short* __restrict__ Out,
                                 int do_relu) {
    // LDS tiles: 128 rows x 32 bf16 (64B) padded to 80B rows (5 x uint4),
    // double buffered: 2 x (10KB A + 10KB B) = 40KB.
    __shared__ uint4 ldsA[2][128 * 5];
    __shared__ uint4 ldsB[2][128 * 5];

    const int t    = threadIdx.x;
    const int n0   = blockIdx.x * 128;     // fast dim = N for L2 reuse of A
    const int m0   = blockIdx.y * 128;
    const int lane = t & 31;
    const int wid  = t >> 5;
    const int wm   = wid & 3;              // wave covers rows wm*32..+31
    const int wn   = wid >> 2;             // wave covers cols wn*64..+63
    const int hl   = lane >> 4;            // lane half (ISA 16-bit layout)
    const int lm   = lane & 15;

    v8f acc[2][4];
    #pragma unroll
    for (int i = 0; i < 2; ++i)
        #pragma unroll
        for (int j = 0; j < 4; ++j) {
            v8f z = {};
            acc[i][j] = z;
        }

    const int lr = t >> 2;                 // 0..63  (row pair loader)
    const int lc = t & 3;                  // 0..3   (16-byte chunk in row)

    // Issue this thread's 4 async B128 loads (2 A rows + 2 B rows) for one
    // K-step into LDS buffer `buf`. No VGPR data path: ASYNCcnt tracked.
    auto issue_tile = [&](int kt, int buf) {
        #pragma unroll
        for (int rr = 0; rr < 2; ++rr) {
            int r = lr + rr * 64;
            const unsigned short* ga = A  + (size_t)(m0 + r) * DH + kt + lc * 8;
            const unsigned short* gb = WT + (size_t)(n0 + r) * DH + kt + lc * 8;
            unsigned la = lds_off(&ldsA[buf][r * 5 + lc]);
            unsigned lb = lds_off(&ldsB[buf][r * 5 + lc]);
            asm volatile(
                "global_load_async_to_lds_b128 %0, %1, off\n\t"
                "global_load_async_to_lds_b128 %2, %3, off"
                :
                : "v"(la), "v"(ga), "v"(lb), "v"(gb)
                : "memory");
        }
    };

    issue_tile(0, 0);

    int buf = 0;
    for (int kt = 0; kt < DH; kt += 32, buf ^= 1) {
        // My async loads for this buffer are done; barrier makes every
        // wave's loads visible and retires all reads of the other buffer.
        asm volatile("s_wait_asynccnt 0x0" ::: "memory");
        __syncthreads();

        if (kt + 32 < DH) issue_tile(kt + 32, buf ^ 1);

        // A fragment: lane half hl -> VGPR0-3 = K hl*8..+7 (bytes hl*16),
        //             VGPR4-7 = K 16+hl*8..+7 (bytes 32+hl*16)
        v16bf af[2], bfr[4];
        #pragma unroll
        for (int mi = 0; mi < 2; ++mi) {
            int row = wm * 32 + mi * 16 + lm;
            af[mi] = make_frag(ldsA[buf][row * 5 + hl],
                               ldsA[buf][row * 5 + 2 + hl]);
        }
        #pragma unroll
        for (int ni = 0; ni < 4; ++ni) {
            int row = wn * 64 + ni * 16 + lm;
            bfr[ni] = make_frag(ldsB[buf][row * 5 + hl],
                                ldsB[buf][row * 5 + 2 + hl]);
        }
        #pragma unroll
        for (int mi = 0; mi < 2; ++mi)
            #pragma unroll
            for (int ni = 0; ni < 4; ++ni)
                acc[mi][ni] = __builtin_amdgcn_wmma_f32_16x16x32_bf16(
                    false, af[mi], false, bfr[ni],
                    (short)0, acc[mi][ni], false, false);
    }

    // C/D layout: VGPR v holds row m = v + 8*hl, col n = lane%16
    #pragma unroll
    for (int mi = 0; mi < 2; ++mi) {
        #pragma unroll
        for (int ni = 0; ni < 4; ++ni) {
            int n = n0 + wn * 64 + ni * 16 + lm;
            #pragma unroll
            for (int v = 0; v < 8; ++v) {
                int m = m0 + wm * 32 + mi * 16 + hl * 8 + v;
                float f = acc[mi][ni][v];
                if (do_relu) f = f > 0.f ? f : 0.f;
                Out[(size_t)m * DH + n] = f32_to_bf16(f);
            }
        }
    }
}

// ---------------------------------------------------------------------------
// Kernel 4: diagonal linear recurrence, in place over xb (bf16):
//   h_s = a[h] * h_{s-1} + xb[s];  xb[s] <- h_s
// One thread per (batch, channel) pair: 131072 chains of 784 steps,
// fully coalesced (256 KB contiguous per step).
// ---------------------------------------------------------------------------
__launch_bounds__(256)
__global__ void scan_kernel(unsigned short* __restrict__ xb,
                            const float* __restrict__ a_l) {
    int idx = blockIdx.x * 256 + threadIdx.x;       // 0..131071
    int h = idx & 511;
    float av  = a_l[h];
    float acc = 0.f;
    unsigned short* p = xb + idx;
    for (int s = 0; s < SEQ_LEN; ++s) {
        acc = fmaf(acc, av, bf16_to_f32(*p));
        *p = f32_to_bf16(acc);
        p += BATCH * DH;
    }
}

// ---------------------------------------------------------------------------
// Kernel 5: out[m] = dot(act[m,:], Wf) + bf + x[m]   (one wave32 per row)
// ---------------------------------------------------------------------------
__launch_bounds__(256)
__global__ void final_kernel(const unsigned short* __restrict__ act,
                             const float* __restrict__ Wf,
                             const float* __restrict__ bias,
                             const float* __restrict__ x,
                             float* __restrict__ out) {
    int gw   = (blockIdx.x * 256 + threadIdx.x) >> 5;   // row index
    int lane = threadIdx.x & 31;
    const unsigned short* row = act + (size_t)gw * DH;
    float s = 0.f;
    #pragma unroll
    for (int i = 0; i < 16; ++i) {
        int k = lane + i * 32;
        s = fmaf(bf16_to_f32(row[k]), Wf[k], s);
    }
    #pragma unroll
    for (int off = 16; off > 0; off >>= 1)
        s += __shfl_xor(s, off, 32);
    if (lane == 0) out[gw] = s + bias[0] + x[gw];
}

// ---------------------------------------------------------------------------
// Host-side orchestration.
// Workspace layout (bytes):
//   [0, 205520896)            act  (bf16, MROWS x 512)
//   [205520896, 411041792)    xb/hs (bf16, MROWS x 512, scanned in place)
//   [411041792, 414187520)    WT   (bf16, 6 x 512 x 512, [2l+isC][n][k])
// ---------------------------------------------------------------------------
extern "C" void kernel_launch(void* const* d_in, const int* in_sizes, int n_in,
                              void* d_out, int out_size, void* d_ws, size_t ws_size,
                              hipStream_t stream) {
    const float* x  = (const float*)d_in[0];
    const float* W0 = (const float*)d_in[1];
    const float* b0 = (const float*)d_in[2];
    const float* a  = (const float*)d_in[3];
    const float* B  = (const float*)d_in[4];
    const float* C  = (const float*)d_in[5];
    const float* Wf = (const float*)d_in[6];
    const float* bf = (const float*)d_in[7];
    float* out = (float*)d_out;

    const size_t ACT_BYTES = (size_t)MROWS * DH * 2;          // 205,520,896
    unsigned short* act = (unsigned short*)d_ws;
    unsigned short* xb  = (unsigned short*)((char*)d_ws + ACT_BYTES);
    unsigned short* wt  = (unsigned short*)((char*)d_ws + 2 * ACT_BYTES);

    // 1. weights -> bf16 transposed fragments
    wprep_kernel<<<(6 * DH * DH) / 256, 256, 0, stream>>>(B, C, wt);

    // 2. input projection + relu -> bf16 act
    inproj_kernel<<<((size_t)MROWS * DH) / 256, 256, 0, stream>>>(x, W0, b0, act);

    // 3. layers: GEMM -> scan -> GEMM(+relu)
    dim3 ggrid(DH / 128, MROWS / 128);   // N fast for L2 reuse of A tiles
    for (int l = 0; l < LAYERS; ++l) {
        gemm_bf16_kernel<<<ggrid, 256, 0, stream>>>(
            act, wt + (size_t)(2 * l) * DH * DH, xb, 0);
        scan_kernel<<<(BATCH * DH) / 256, 256, 0, stream>>>(xb, a + l * DH);
        gemm_bf16_kernel<<<ggrid, 256, 0, stream>>>(
            xb, wt + (size_t)(2 * l + 1) * DH * DH, act, 1);
    }

    // 4. output projection + bias + residual
    final_kernel<<<(MROWS * 32) / 256, 256, 0, stream>>>(act, Wf, bf, x, out);
}
